// lstm_autoencoder_15298673509092
// MI455X (gfx1250) — matrix operand order repaired and checked by reading, not compiled, not run on hardware
//
#include <hip/hip_runtime.h>

// ---------------------------------------------------------------------------
// LSTM autoencoder, MI455X (gfx1250, wave32) plan:
//   K1: XG[T][B][32](f16) = x[B*T,32] @ e1_Wih^T + (e1_bih+e1_bhh)   (WMMA f16)
//   K2: per-lane fused recurrence e1->e2->bridge->d1->d2, weights staged
//       into LDS with GLOBAL_LOAD_ASYNC_TO_LDS_B32 (ASYNCcnt), writes
//       H4[B][T][8](f16)
//   K3: out[B*T,32](f32) = H4 @ out_W^T + out_b                      (WMMA f16)
// Workspace: 64MB (XG) + 16MB (H4) = 80MB of d_ws.
// ---------------------------------------------------------------------------

typedef __attribute__((ext_vector_type(16))) _Float16 v16h;
typedef __attribute__((ext_vector_type(8)))  _Float16 v8h;
typedef __attribute__((ext_vector_type(8)))  float    v8f;

#define B_ 4096
#define T_ 256
#define I_ 32

__device__ __forceinline__ float sigf(float x) {
    return 1.0f / (1.0f + __expf(-x));
}
__device__ __forceinline__ float tanh_fast(float x) {
    x = fminf(fmaxf(x, -15.0f), 15.0f);
    float e = __expf(2.0f * x);
    return (e - 1.0f) / (e + 1.0f);
}
__device__ __forceinline__ void cvt4(v16h& a, int base, float4 p) {
    a[base + 0] = (_Float16)p.x;
    a[base + 1] = (_Float16)p.y;
    a[base + 2] = (_Float16)p.z;
    a[base + 3] = (_Float16)p.w;
}

// ---- async global->LDS copy (CDNA5 ASYNCcnt path) -------------------------
// VGLOBAL async encoding: VDST VGPR holds the LDS byte address,
// VADDR the 64-bit global address (GV mode, SADDR=off).
__device__ __forceinline__ unsigned ldsoff(const void* p) {
    return (unsigned)(uintptr_t)p;   // low 32 bits of flat addr = LDS offset
}
__device__ __forceinline__ void async_copy_b32(const float* gsrc, unsigned lds_byte_off) {
    asm volatile("global_load_async_to_lds_b32 %0, %1, off"
                 :: "v"(lds_byte_off), "v"(gsrc)
                 : "memory");
}
__device__ __forceinline__ void wait_async0() {
#if __has_builtin(__builtin_amdgcn_s_wait_asynccnt)
    __builtin_amdgcn_s_wait_asynccnt(0);
#else
    asm volatile("s_wait_asynccnt 0x0" ::: "memory");
#endif
}

// ---------------------------------------------------------------------------
// Kernel 1: e1 input GEMM.  Each wave: 16 rows (natural b*T+t order) x 32 cols.
// A fragment (16x32 f16): lane m=l&15, half=l>>4; elem e -> K = (e<8 ? e : e+8) + 8*half
// B fragment (32x16 f16): lane n=l&15; elem e -> K = (e<8 ? e : e+8) + 8*half;  B[k][n] = Wih[n][k]
// D (16x16 f32): lane col N = l&15, VGPR i -> row M = i + 8*half
// ---------------------------------------------------------------------------
__global__ void __launch_bounds__(256)
xgemm_e1_kernel(const float* __restrict__ x,
                const float* __restrict__ Wih,
                const float* __restrict__ bih,
                const float* __restrict__ bhh,
                _Float16* __restrict__ XG)   // [T][B][32] f16
{
    const int lane = threadIdx.x & 31;
    const int wave = threadIdx.x >> 5;
    const int tile = blockIdx.x * 8 + wave;   // 16-row tile over BT
    const int row0 = tile * 16;               // row index r = b*T + t
    const int half = lane >> 4;
    const int mrow = lane & 15;

    // A fragment: contiguous x reads (2KB per wave)
    const float4* af = (const float4*)(x + (size_t)(row0 + mrow) * I_);
    v16h a;
    cvt4(a, 0,  af[half * 2 + 0]);
    cvt4(a, 4,  af[half * 2 + 1]);
    cvt4(a, 8,  af[half * 2 + 4]);
    cvt4(a, 12, af[half * 2 + 5]);

    const int b_idx = row0 >> 8;        // / T_
    const int t0    = row0 & (T_ - 1);

#pragma unroll
    for (int nt = 0; nt < 2; ++nt) {
        const int N = nt * 16 + mrow;
        const float4* wf = (const float4*)(Wih + N * I_ + half * 8);
        v16h bfrag;
        cvt4(bfrag, 0,  wf[0]);
        cvt4(bfrag, 4,  wf[1]);
        cvt4(bfrag, 8,  wf[4]);
        cvt4(bfrag, 12, wf[5]);

        v8f c = {};
        c = __builtin_amdgcn_wmma_f32_16x16x32_f16(false, a, false, bfrag,
                                                   (short)0, c, false, false);
        const float bias = bih[N] + bhh[N];
#pragma unroll
        for (int i = 0; i < 8; ++i) {
            const int M = i + half * 8;
            const int t = t0 + M;
            XG[((size_t)t * B_ + b_idx) * I_ + N] = (_Float16)(c[i] + bias);
        }
    }
}

// ---------------------------------------------------------------------------
// Kernel 2: fused recurrence. One lane == one batch row; weights in LDS
// (staged via async-to-LDS DMA).
// ---------------------------------------------------------------------------
__global__ void __launch_bounds__(256)
lstm_recurrent_kernel(const _Float16* __restrict__ XG,   // [T][B][32]
                      const float* __restrict__ W1hh,    // [32][8]
                      const float* __restrict__ W2ih,    // [16][8]
                      const float* __restrict__ W2hh,    // [16][4]
                      const float* __restrict__ b2ih,
                      const float* __restrict__ b2hh,
                      const float* __restrict__ brW,     // [4][4]
                      const float* __restrict__ brb,     // [4]
                      const float* __restrict__ d1ih,    // [16][4]
                      const float* __restrict__ d1hh,    // [16][4]
                      const float* __restrict__ d1bih,
                      const float* __restrict__ d1bhh,
                      const float* __restrict__ d2ih,    // [32][4]
                      const float* __restrict__ d2hh,    // [32][8]
                      const float* __restrict__ d2bih,
                      const float* __restrict__ d2bhh,
                      _Float16* __restrict__ H4)         // [B][T][8]
{
    __shared__ float sW1hh[256], sW2ih[128], sW2hh[64], sb2[16];
    __shared__ float sbrW[16], sbrb[4];
    __shared__ float sd1ih[64], sd1hh[64], sd1b[16];
    __shared__ float sd2ih[128], sd2hh[256], sd2b[32];

    const int tid = threadIdx.x;

    // -- pure weight copies: async DMA global->LDS (tracked by ASYNCcnt) --
    async_copy_b32(W1hh + tid, ldsoff(&sW1hh[tid]));                 // 256
    async_copy_b32(d2hh + tid, ldsoff(&sd2hh[tid]));                 // 256
    if (tid < 128) {
        async_copy_b32(W2ih + tid, ldsoff(&sW2ih[tid]));             // 128
        async_copy_b32(d2ih + tid, ldsoff(&sd2ih[tid]));             // 128
    }
    if (tid < 64) {
        async_copy_b32(W2hh + tid, ldsoff(&sW2hh[tid]));             // 64
        async_copy_b32(d1ih + tid, ldsoff(&sd1ih[tid]));             // 64
        async_copy_b32(d1hh + tid, ldsoff(&sd1hh[tid]));             // 64
    }
    if (tid < 16) async_copy_b32(brW + tid, ldsoff(&sbrW[tid]));     // 16

    // -- combined biases need arithmetic: regular path --
    if (tid < 32) sd2b[tid] = d2bih[tid] + d2bhh[tid];
    if (tid < 16) { sb2[tid]  = b2ih[tid] + b2hh[tid];
                    sd1b[tid] = d1bih[tid] + d1bhh[tid]; }
    if (tid < 4)  sbrb[tid] = brb[tid];

    wait_async0();
    __syncthreads();

    const int b = blockIdx.x * blockDim.x + tid;   // one lane per batch row

    float h1[8] = {}, c1[8] = {}, h2[4] = {}, c2[4] = {};

    // ---------------- encoder ----------------
    for (int t = 0; t < T_; ++t) {
        const v16h* xgv = (const v16h*)(XG + ((size_t)t * B_ + b) * I_);
        v16h x0 = xgv[0], x1 = xgv[1];
        if (t + 1 < T_)
            __builtin_prefetch(XG + ((size_t)(t + 1) * B_ + b) * I_, 0, 1);

        float g[32];
#pragma unroll
        for (int e = 0; e < 16; ++e) { g[e] = (float)x0[e]; g[16 + e] = (float)x1[e]; }
#pragma unroll
        for (int k = 0; k < 32; ++k) {
            float acc = g[k];
#pragma unroll
            for (int j = 0; j < 8; ++j) acc += sW1hh[k * 8 + j] * h1[j];
            g[k] = acc;
        }
#pragma unroll
        for (int j = 0; j < 8; ++j) {   // gate order i,f,g,o
            c1[j] = sigf(g[8 + j]) * c1[j] + sigf(g[j]) * tanh_fast(g[16 + j]);
            h1[j] = sigf(g[24 + j]) * tanh_fast(c1[j]);
        }
        float g2[16];
#pragma unroll
        for (int k = 0; k < 16; ++k) {
            float acc = sb2[k];
#pragma unroll
            for (int j = 0; j < 8; ++j) acc += sW2ih[k * 8 + j] * h1[j];
#pragma unroll
            for (int j = 0; j < 4; ++j) acc += sW2hh[k * 4 + j] * h2[j];
            g2[k] = acc;
        }
#pragma unroll
        for (int j = 0; j < 4; ++j) {
            c2[j] = sigf(g2[4 + j]) * c2[j] + sigf(g2[j]) * tanh_fast(g2[8 + j]);
            h2[j] = sigf(g2[12 + j]) * tanh_fast(c2[j]);
        }
    }

    // ---------------- bridge ----------------
    float z[4];
#pragma unroll
    for (int i = 0; i < 4; ++i) {
        float acc = sbrb[i];
#pragma unroll
        for (int j = 0; j < 4; ++j) acc += sbrW[i * 4 + j] * h2[j];
        z[i] = acc;
    }
    float zg1[16];   // d1 input gates are time-invariant
#pragma unroll
    for (int k = 0; k < 16; ++k) {
        float acc = sd1b[k];
#pragma unroll
        for (int j = 0; j < 4; ++j) acc += sd1ih[k * 4 + j] * z[j];
        zg1[k] = acc;
    }

    // ---------------- decoder ----------------
    float h3[4] = {}, c3[4] = {}, h4[8] = {}, c4[8] = {};
    _Float16* hrow = H4 + (size_t)b * T_ * 8;
    for (int t = 0; t < T_; ++t) {
        float g3[16];
#pragma unroll
        for (int k = 0; k < 16; ++k) {
            float acc = zg1[k];
#pragma unroll
            for (int j = 0; j < 4; ++j) acc += sd1hh[k * 4 + j] * h3[j];
            g3[k] = acc;
        }
#pragma unroll
        for (int j = 0; j < 4; ++j) {
            c3[j] = sigf(g3[4 + j]) * c3[j] + sigf(g3[j]) * tanh_fast(g3[8 + j]);
            h3[j] = sigf(g3[12 + j]) * tanh_fast(c3[j]);
        }
        float g4[32];
#pragma unroll
        for (int k = 0; k < 32; ++k) {
            float acc = sd2b[k];
#pragma unroll
            for (int j = 0; j < 4; ++j) acc += sd2ih[k * 4 + j] * h3[j];
#pragma unroll
            for (int j = 0; j < 8; ++j) acc += sd2hh[k * 8 + j] * h4[j];
            g4[k] = acc;
        }
#pragma unroll
        for (int j = 0; j < 8; ++j) {
            c4[j] = sigf(g4[8 + j]) * c4[j] + sigf(g4[j]) * tanh_fast(g4[16 + j]);
            h4[j] = sigf(g4[24 + j]) * tanh_fast(c4[j]);
        }
        v8h hv;
#pragma unroll
        for (int j = 0; j < 8; ++j) hv[j] = (_Float16)h4[j];
        *(v8h*)(hrow + t * 8) = hv;   // 16B store
    }
}

// ---------------------------------------------------------------------------
// Kernel 3: output Linear(8->32) via WMMA, K=8 zero-padded to 32.
// Only half==0 lanes, elems 0..7 carry real K (K<8); everything else is 0.
// ---------------------------------------------------------------------------
__global__ void __launch_bounds__(256)
out_gemm_kernel(const _Float16* __restrict__ H4,  // [B*T][8]
                const float* __restrict__ outW,   // [32][8]
                const float* __restrict__ outb,   // [32]
                float* __restrict__ out)          // [B*T][32]
{
    const int lane = threadIdx.x & 31;
    const int wave = threadIdx.x >> 5;
    const int tile = blockIdx.x * 8 + wave;
    const int row0 = tile * 16;
    const int half = lane >> 4;
    const int mrow = lane & 15;

    v16h a = {};
    if (half == 0) {
        v8h ar = *(const v8h*)(H4 + (size_t)(row0 + mrow) * 8);
#pragma unroll
        for (int e = 0; e < 8; ++e) a[e] = ar[e];
    }

#pragma unroll
    for (int nt = 0; nt < 2; ++nt) {
        const int N = nt * 16 + mrow;
        v16h bfrag = {};
        if (half == 0) {
            const float* wrow = outW + N * 8;
#pragma unroll
            for (int e = 0; e < 8; ++e) bfrag[e] = (_Float16)wrow[e];
        }
        v8f c = {};
        c = __builtin_amdgcn_wmma_f32_16x16x32_f16(false, a, false, bfrag,
                                                   (short)0, c, false, false);
        const float bias = outb[N];
#pragma unroll
        for (int i = 0; i < 8; ++i) {
            const int M = i + half * 8;
            out[(size_t)(row0 + M) * 32 + N] = c[i] + bias;
        }
    }
}

// ---------------------------------------------------------------------------
extern "C" void kernel_launch(void* const* d_in, const int* in_sizes, int n_in,
                              void* d_out, int out_size, void* d_ws, size_t ws_size,
                              hipStream_t stream)
{
    (void)in_sizes; (void)n_in; (void)out_size; (void)ws_size;

    const float* x      = (const float*)d_in[0];
    const float* e1_Wih = (const float*)d_in[1];
    const float* e1_Whh = (const float*)d_in[2];
    const float* e1_bih = (const float*)d_in[3];
    const float* e1_bhh = (const float*)d_in[4];
    const float* e2_Wih = (const float*)d_in[5];
    const float* e2_Whh = (const float*)d_in[6];
    const float* e2_bih = (const float*)d_in[7];
    const float* e2_bhh = (const float*)d_in[8];
    const float* br_W   = (const float*)d_in[9];
    const float* br_b   = (const float*)d_in[10];
    const float* d1_Wih = (const float*)d_in[11];
    const float* d1_Whh = (const float*)d_in[12];
    const float* d1_bih = (const float*)d_in[13];
    const float* d1_bhh = (const float*)d_in[14];
    const float* d2_Wih = (const float*)d_in[15];
    const float* d2_Whh = (const float*)d_in[16];
    const float* d2_bih = (const float*)d_in[17];
    const float* d2_bhh = (const float*)d_in[18];
    const float* out_W  = (const float*)d_in[19];
    const float* out_b  = (const float*)d_in[20];

    const size_t BT = (size_t)B_ * T_;
    _Float16* XG = (_Float16*)d_ws;                                       // 64 MB
    _Float16* H4 = (_Float16*)((char*)d_ws + BT * 32 * sizeof(_Float16)); // +16 MB

    dim3 blk(256);
    // 8 waves/block, 16 rows/wave -> 128 rows per block
    xgemm_e1_kernel<<<(int)(BT / 128), blk, 0, stream>>>(x, e1_Wih, e1_bih, e1_bhh, XG);

    lstm_recurrent_kernel<<<B_ / 256, blk, 0, stream>>>(
        XG, e1_Whh, e2_Wih, e2_Whh, e2_bih, e2_bhh,
        br_W, br_b, d1_Wih, d1_Whh, d1_bih, d1_bhh,
        d2_Wih, d2_Whh, d2_bih, d2_bhh, H4);

    out_gemm_kernel<<<(int)(BT / 128), blk, 0, stream>>>(H4, out_W, out_b, (float*)d_out);
}
